// VQMMApool_44573170598174
// MI455X (gfx1250) — matrix operand order repaired and checked by reading, not compiled, not run on hardware
//
#include <hip/hip_runtime.h>
#include <cstdint>

#define NEG_INF_F (-1e30f)

constexpr int B_   = 8;
constexpr int LV   = 256;
constexpr int LQ   = 64;
constexpr int D_   = 768;   // reduction (d) dimension
constexpr int KOUT = 768;   // hidden (k) dimension
constexpr int DOUT = 2304;  // output feature dim (3*D)
constexpr int LDA  = 776;   // padded LDS row stride in bf16 (16B aligned, bank-spread)

typedef __attribute__((ext_vector_type(16))) __bf16 v16bf;
typedef __attribute__((ext_vector_type(8)))  __bf16 v8bf;
typedef __attribute__((ext_vector_type(8)))  float  v8f;

// float -> bf16 round-to-nearest-even (pure integer ops)
__device__ inline __bf16 f2bf(float x) {
  union { float f; uint32_t u; } a; a.f = x;
  uint32_t r = a.u + 0x7FFFu + ((a.u >> 16) & 1u);
  union { unsigned short s; __bf16 b; } o; o.s = (unsigned short)(r >> 16);
  return o.b;
}

__device__ inline v16bf combine8(v8bf lo, v8bf hi) {
  return __builtin_shufflevector(lo, hi, 0,1,2,3,4,5,6,7,8,9,10,11,12,13,14,15);
}

// A fragment (16x32 bf16, M x K): lane<16 -> row M=lane, K = {d0+0..7, d0+16..23};
// lane>=16 -> row M=lane-16, K = {d0+8..15, d0+24..31}. Two contiguous 16B LDS loads.
__device__ inline v16bf load_afrag(const __bf16* As, int d0, int half, int lidx) {
  const __bf16* p = As + lidx * LDA + d0 + half * 8;
  v8bf lo = *(const v8bf*)p;
  v8bf hi = *(const v8bf*)(p + 16);
  return combine8(lo, hi);
}

// B fragment (32x16 bf16, K x N): lane -> column N=k0+(lane&15); K = {d0+8h+0..7, d0+16+8h+0..7}, h=lane>>4.
// WT is d-major (D x KOUT) bf16 -> strided scalar loads per lane.
__device__ inline v16bf load_bfrag(const __bf16* WT, int d0, int k0, int lane) {
  const __bf16* p = WT + (size_t)(d0 + (lane >> 4) * 8) * KOUT + k0 + (lane & 15);
  v16bf r;
  #pragma unroll
  for (int s = 0; s < 8; ++s) {
    r[s] = p[(size_t)s * KOUT];
    r[8 + s] = p[(size_t)(16 + s) * KOUT];
  }
  return r;
}

// ---------------- W1 slices -> d-major bf16 (WvT | WqT | WmT) ----------------
__global__ void prep_wt_kernel(const float* __restrict__ W1, __bf16* __restrict__ WT) {
  int idx = blockIdx.x * 256 + threadIdx.x;
  if (idx >= 3 * D_ * KOUT) return;
  int s = idx / (D_ * KOUT);
  int r = idx - s * (D_ * KOUT);
  int d = r / KOUT;
  int k = r - d * KOUT;
  WT[idx] = f2bf(W1[(size_t)k * (3 * D_) + s * D_ + d]);   // WT[s][d][k] = W1[k][s*768+d]
}

// ---------------- 16-row x 768-col bf16 WMMA GEMM: Y = X * W^T (N-blocked x4) ----------------
__global__ void __launch_bounds__(128) gemm16_bf16(
    const float* __restrict__ X, const __bf16* __restrict__ WT, float* __restrict__ Y)
{
  __shared__ __bf16 As[16 * LDA];
  const int row0 = blockIdx.x * 16;
  const int tid = threadIdx.x;
  for (int idx = tid; idx < 16 * D_; idx += 128) {
    int r = idx / D_;
    int d = idx - r * D_;
    As[r * LDA + d] = f2bf(X[(size_t)(row0 + r) * D_ + d]);
  }
  __syncthreads();

  const int lane = tid & 31, wave = tid >> 5;
  const int half = lane >> 4, lidx = lane & 15;

  for (int g = 0; g < 3; ++g) {                 // wave owns 12 contiguous k-tiles, 3 groups of 4
    const int kt0 = wave * 12 + g * 4;
    v8f acc0 = {}, acc1 = {}, acc2 = {}, acc3 = {};
    #pragma unroll 2
    for (int dt = 0; dt < 24; ++dt) {
      const int d0 = dt * 32;
      v16bf af = load_afrag(As, d0, half, lidx);
      v16bf b0 = load_bfrag(WT, d0, (kt0 + 0) * 16, lane);
      v16bf b1 = load_bfrag(WT, d0, (kt0 + 1) * 16, lane);
      v16bf b2 = load_bfrag(WT, d0, (kt0 + 2) * 16, lane);
      v16bf b3 = load_bfrag(WT, d0, (kt0 + 3) * 16, lane);
      acc0 = __builtin_amdgcn_wmma_f32_16x16x32_bf16(false, af, false, b0, (short)0, acc0, false, false);
      acc1 = __builtin_amdgcn_wmma_f32_16x16x32_bf16(false, af, false, b1, (short)0, acc1, false, false);
      acc2 = __builtin_amdgcn_wmma_f32_16x16x32_bf16(false, af, false, b2, (short)0, acc2, false, false);
      acc3 = __builtin_amdgcn_wmma_f32_16x16x32_bf16(false, af, false, b3, (short)0, acc3, false, false);
    }
    // C layout: lane<16: VGPR r -> (M=r, N=lidx); lane>=16: (M=8+r, N=lidx)
    #pragma unroll
    for (int r = 0; r < 8; ++r) {
      const int m = half * 8 + r;
      float* yrow = Y + (size_t)(row0 + m) * KOUT + lidx;
      yrow[(kt0 + 0) * 16] = acc0[r];
      yrow[(kt0 + 1) * 16] = acc1[r];
      yrow[(kt0 + 2) * 16] = acc2[r];
      yrow[(kt0 + 3) * 16] = acc3[r];
    }
  }
}

// ------- hot kernel: block = (b,v) x all 64 q; wave = 16-q strip, all 48 k-tiles -------
__global__ void __launch_bounds__(128) pair_logits_kernel(
    const float* __restrict__ vid, const float* __restrict__ query,
    const float* __restrict__ vmask, const float* __restrict__ qmask,
    const float* __restrict__ a_ws, const float* __restrict__ bq_ws,
    const __bf16* __restrict__ WmT, const float* __restrict__ W2,
    float* __restrict__ logits)
{
  __shared__ __bf16 As[64 * LDA];        // 64 pair rows x 768 d, bf16 (~97 KB)

  const int bid = blockIdx.x;            // B*LV blocks
  const int v = bid & (LV - 1);
  const int b = bid >> 8;
  const int tid = threadIdx.x;

  // stage A: row q = bf16( vid[b,v,:] * query[b,q,:] ), q = 0..63
  const float* vrow = vid + (size_t)(b * LV + v) * D_;
  for (int idx = tid; idx < 64 * D_; idx += 128) {
    int r = idx / D_;
    int d = idx - r * D_;
    float qv = query[(size_t)(b * LQ + r) * D_ + d];
    As[r * LDA + d] = f2bf(vrow[d] * qv);
  }
  __syncthreads();

  const int lane = tid & 31, wave = tid >> 5;
  const int half = lane >> 4, lidx = lane & 15;
  const __bf16* Aw = As + wave * 16 * LDA;     // this wave's 16 pair rows (q = wave*16 + m)

  float simacc[8];
  #pragma unroll
  for (int r = 0; r < 8; ++r) simacc[r] = 0.f;

  const size_t arow = (size_t)(b * LV + v) * KOUT;

  for (int g = 0; g < 12; ++g) {               // 48 k-tiles in groups of 4 (A reused x4)
    const int kt0 = g * 4;
    v8f acc0 = {}, acc1 = {}, acc2 = {}, acc3 = {};
    #pragma unroll 2
    for (int dt = 0; dt < 24; ++dt) {          // full d reduction (768)
      const int d0 = dt * 32;
      v16bf af = load_afrag(Aw, d0, half, lidx);
      v16bf b0 = load_bfrag(WmT, d0, (kt0 + 0) * 16, lane);  // 4 frags = 128 contiguous B per d row
      v16bf b1 = load_bfrag(WmT, d0, (kt0 + 1) * 16, lane);
      v16bf b2 = load_bfrag(WmT, d0, (kt0 + 2) * 16, lane);
      v16bf b3 = load_bfrag(WmT, d0, (kt0 + 3) * 16, lane);
      acc0 = __builtin_amdgcn_wmma_f32_16x16x32_bf16(false, af, false, b0, (short)0, acc0, false, false);
      acc1 = __builtin_amdgcn_wmma_f32_16x16x32_bf16(false, af, false, b1, (short)0, acc1, false, false);
      acc2 = __builtin_amdgcn_wmma_f32_16x16x32_bf16(false, af, false, b2, (short)0, acc2, false, false);
      acc3 = __builtin_amdgcn_wmma_f32_16x16x32_bf16(false, af, false, b3, (short)0, acc3, false, false);
    }
    // fuse: h = relu(m + a_k + b_k); sim += h * w2_k  (lane owns k column k0+lidx)
    #pragma unroll
    for (int j = 0; j < 4; ++j) {
      const int k = (kt0 + j) * 16 + lidx;
      const float a_val = a_ws[arow + k];
      const float w2v = W2[k];
      const v8f* accp = (j == 0) ? &acc0 : (j == 1) ? &acc1 : (j == 2) ? &acc2 : &acc3;
      #pragma unroll
      for (int r = 0; r < 8; ++r) {
        const int q = wave * 16 + half * 8 + r;
        const float bqv = bq_ws[(size_t)(b * LQ + q) * KOUT + k];
        float h = (*accp)[r] + a_val + bqv;
        h = h > 0.f ? h : 0.f;
        simacc[r] = fmaf(h, w2v, simacc[r]);
      }
    }
  }

  // reduce over the 16 k-lanes of each half-wave; each wave fully owns its 16 rows
  #pragma unroll
  for (int r = 0; r < 8; ++r) {
    float s = simacc[r];
    s += __shfl_xor(s, 1, 32);
    s += __shfl_xor(s, 2, 32);
    s += __shfl_xor(s, 4, 32);
    s += __shfl_xor(s, 8, 32);
    if (lidx == 0) {
      const int q = wave * 16 + half * 8 + r;
      const float pm = vmask[b * LV + v] * qmask[b * LQ + q];
      logits[(size_t)(b * LV + v) * LQ + q] = s + (1.f - pm) * NEG_INF_F;
    }
  }
}

// ---------------- q2v: rowmax over q, softmax over v, Q2V[b,d] ----------------
__global__ void __launch_bounds__(256) q2v_kernel(
    const float* __restrict__ logits, const float* __restrict__ vid,
    float* __restrict__ Q2V)
{
  __shared__ float sm[256];
  __shared__ float att[256];
  const int b = blockIdx.x;
  const int v = threadIdx.x;
  const float* lrow = logits + (size_t)(b * LV + v) * LQ;
  float mx = -3.4e38f;
  for (int q = 0; q < LQ; ++q) mx = fmaxf(mx, lrow[q]);
  sm[v] = mx;
  __syncthreads();
  for (int s = 128; s > 0; s >>= 1) { if (v < s) sm[v] = fmaxf(sm[v], sm[v + s]); __syncthreads(); }
  const float MX = sm[0];
  __syncthreads();
  const float e = __expf(mx - MX);
  sm[v] = e;
  __syncthreads();
  for (int s = 128; s > 0; s >>= 1) { if (v < s) sm[v] += sm[v + s]; __syncthreads(); }
  const float SUM = sm[0];
  att[v] = e / SUM;
  __syncthreads();
  for (int d = v; d < D_; d += 256) {
    float acc = 0.f;
    for (int vv = 0; vv < LV; ++vv)
      acc = fmaf(att[vv], vid[(size_t)(b * LV + vv) * D_ + d], acc);
    Q2V[b * D_ + d] = acc;
  }
}

// ---------------- v2q softmax, V2Q matvec, final output assembly ----------------
__global__ void __launch_bounds__(256) v2q_out_kernel(
    const float* __restrict__ logits, const float* __restrict__ vid,
    const float* __restrict__ query, const float* __restrict__ Q2V,
    float* __restrict__ out)
{
  __shared__ float att[LQ];
  __shared__ float red[LQ];
  const int bid = blockIdx.x;
  const int b = bid >> 8;
  const int v = bid & (LV - 1);
  const int tid = threadIdx.x;
  float lv = 0.f;
  if (tid < LQ) { lv = logits[(size_t)(b * LV + v) * LQ + tid]; red[tid] = lv; }
  __syncthreads();
  for (int s = 32; s > 0; s >>= 1) { if (tid < s) red[tid] = fmaxf(red[tid], red[tid + s]); __syncthreads(); }
  const float MX = red[0];
  __syncthreads();
  float e = 0.f;
  if (tid < LQ) { e = __expf(lv - MX); red[tid] = e; }
  __syncthreads();
  for (int s = 32; s > 0; s >>= 1) { if (tid < s) red[tid] += red[tid + s]; __syncthreads(); }
  const float SUM = red[0];
  if (tid < LQ) att[tid] = e / SUM;
  __syncthreads();

  const size_t rowi = (size_t)(b * LV + v);
  float* orow = out + rowi * DOUT;
  for (int d = tid; d < D_; d += 256) {
    float acc = 0.f;
    for (int q = 0; q < LQ; ++q)
      acc = fmaf(att[q], query[(size_t)(b * LQ + q) * D_ + d], acc);
    const float vf = vid[rowi * D_ + d];
    orow[d]          = vf;                       // vid_feat
    orow[D_ + d]     = acc;                      // V2Q
    orow[2 * D_ + d] = vf * Q2V[b * D_ + d];     // Q2V2Q
  }
}

extern "C" void kernel_launch(void* const* d_in, const int* in_sizes, int n_in,
                              void* d_out, int out_size, void* d_ws, size_t ws_size,
                              hipStream_t stream)
{
  (void)in_sizes; (void)n_in; (void)out_size; (void)ws_size;
  const float* vid   = (const float*)d_in[0];
  const float* query = (const float*)d_in[1];
  const float* vmask = (const float*)d_in[2];
  const float* qmask = (const float*)d_in[3];
  const float* W1    = (const float*)d_in[4];
  const float* W2    = (const float*)d_in[5];
  float* out = (float*)d_out;

  // workspace layout (~12 MB total)
  char* ws = (char*)d_ws;
  __bf16* WT = (__bf16*)ws;                                  // 3*768*768 bf16 = 3.5 MB
  size_t off = (size_t)3 * D_ * KOUT * sizeof(__bf16);
  float* a_ws   = (float*)(ws + off); off += (size_t)B_ * LV * KOUT * 4;  // 6.3 MB
  float* bq_ws  = (float*)(ws + off); off += (size_t)B_ * LQ * KOUT * 4;  // 1.6 MB
  float* logits = (float*)(ws + off); off += (size_t)B_ * LV * LQ * 4;    // 0.5 MB
  float* Q2V    = (float*)(ws + off);                                      // 24 KB

  prep_wt_kernel<<<(3 * D_ * KOUT + 255) / 256, 256, 0, stream>>>(W1, WT);
  gemm16_bf16<<<(B_ * LV) / 16, 128, 0, stream>>>(vid, WT, a_ws);
  gemm16_bf16<<<(B_ * LQ) / 16, 128, 0, stream>>>(query, WT + (size_t)D_ * KOUT, bq_ws);
  pair_logits_kernel<<<B_ * LV, 128, 0, stream>>>(
      vid, query, vmask, qmask, a_ws, bq_ws, WT + (size_t)2 * D_ * KOUT, W2, logits);
  q2v_kernel<<<B_, 256, 0, stream>>>(logits, vid, Q2V);
  v2q_out_kernel<<<B_ * LV, 256, 0, stream>>>(logits, vid, query, Q2V, out);
}
